// DenseConv2d_44573170598032
// MI455X (gfx1250) — matrix operand order repaired and checked by reading, not compiled, not run on hardware
//
#include <hip/hip_runtime.h>
#include <stdint.h>

// ---------------------------------------------------------------------------
// Implicit-GEMM 3x3 conv (stride1, pad1) via split-bf16 (hi/lo) WMMA, gfx1250.
//   C[256][100352] = W[256][1152] * im2col(X)[1152][100352]  (+ bias)
// fp32 emulated by 3 bf16 WMMAs (hi*hi + hi*lo + lo*hi), f32 accumulate.
// Conversion hoisted to prepack kernels. Main loop: double-buffered LDS,
// 1 barrier/chunk; A tiles streamed with global_load_async_to_lds_b128
// (ASYNCcnt), B im2col gathers register-staged under the WMMA block.
// ---------------------------------------------------------------------------

typedef __attribute__((ext_vector_type(16))) __bf16 v16bf;
typedef __attribute__((ext_vector_type(8)))  float  v8f;
typedef unsigned short u16;

#define LDA 40            // padded row stride (ushorts) for a 32-wide K tile
#define KTOT 1152         // 128 * 3 * 3
#define HW 3136           // 56*56
#define CHW 401408        // 128*3136
#define KOHW 802816       // 256*3136
#define IN_ELEMS 12845056 // 32*128*56*56
#define W_ELEMS 294912    // 256*1152

__device__ __forceinline__ u16 f32_to_bf16_rne(float x) {
  uint32_t u = __float_as_uint(x);
  uint32_t r = u + 0x7FFFu + ((u >> 16) & 1u);
  return (u16)(r >> 16);
}

__device__ __forceinline__ void split_bf16(float x, u16& hi, u16& lo) {
  u16 h = f32_to_bf16_rne(x);
  float hf = __uint_as_float(((uint32_t)h) << 16);
  hi = h;
  lo = f32_to_bf16_rne(x - hf);
}

// Async global->LDS copy, 16B per lane (CDNA5 GLOBAL_LOAD_ASYNC_TO_LDS_B128,
// tracked by ASYNCcnt). lds_off = low 32 bits of the generic shared pointer
// (== LDS byte address per ISA aperture rule).
__device__ __forceinline__ void async_copy_b128(uint32_t lds_off, const void* g) {
  asm volatile("global_load_async_to_lds_b128 %0, %1, off"
               :: "v"(lds_off), "v"(g) : "memory");
}
__device__ __forceinline__ void wait_async0() {
  asm volatile("s_wait_asynccnt 0x0" ::: "memory");
}
__device__ __forceinline__ uint32_t lds_off32(const void* p) {
  return (uint32_t)(uintptr_t)p;
}

union FragU { v16bf v; uint4 q[2]; };

// A-matrix 16x32 bf16 fragment: lanes 0-15 hold K {0..7,16..23} of row M=lane,
// lanes 16-31 hold K {8..15,24..31} of row M=lane-16.
__device__ __forceinline__ v16bf load_fragA(const u16* row, int lane) {
  const int kb = (lane < 16) ? 0 : 8;
  FragU f;
  f.q[0] = *(const uint4*)(row + kb);
  f.q[1] = *(const uint4*)(row + kb + 16);
  return f.v;
}

// B-matrix 32x16 bf16 fragment ([n][k] in LDS): lanes 0-15 hold K {0..7,16..23} of
// column N=lane, lanes 16-31 hold K {8..15,24..31} of column N=lane-16.
__device__ __forceinline__ v16bf load_fragB(const u16* row, int lane) {
  const int kb = (lane < 16) ? 0 : 8;
  FragU f;
  f.q[0] = *(const uint4*)(row + kb);
  f.q[1] = *(const uint4*)(row + kb + 16);
  return f.v;
}

// ---------------------------------------------------------------------------
// Prepack kernels
// ---------------------------------------------------------------------------
__global__ __launch_bounds__(256)
void prepack_weight(const float* __restrict__ wgt,
                    u16* __restrict__ whi, u16* __restrict__ wlo) {
  int idx = blockIdx.x * 256 + threadIdx.x;   // 1152 blocks -> 294912
  float x = wgt[idx];
  u16 h, l;
  split_bf16(x, h, l);
  whi[idx] = h;
  wlo[idx] = l;
}

__global__ __launch_bounds__(256)
void prepack_input(const float* __restrict__ in, uint32_t* __restrict__ xpk) {
  int idx4 = blockIdx.x * 256 + threadIdx.x;  // 12544 blocks -> 12845056/4
  float4 x = ((const float4*)in)[idx4];
  float vals[4] = {x.x, x.y, x.z, x.w};
  uint4 p;
  uint32_t pk[4];
#pragma unroll
  for (int e = 0; e < 4; ++e) {
    u16 h, l;
    split_bf16(vals[e], h, l);
    pk[e] = ((uint32_t)h << 16) | (uint32_t)l;
  }
  p.x = pk[0]; p.y = pk[1]; p.z = pk[2]; p.w = pk[3];
  ((uint4*)xpk)[idx4] = p;
}

// ---------------------------------------------------------------------------
// Main implicit-GEMM kernel: async A staging + double-buffered LDS pipeline
// ---------------------------------------------------------------------------
__global__ __launch_bounds__(256)
void conv3x3_igemm_pk(const uint32_t* __restrict__ xpk,
                      const u16* __restrict__ whi,
                      const u16* __restrict__ wlo,
                      const float* __restrict__ bias,
                      float* __restrict__ out) {
  __shared__ alignas(16) u16 sAhi0[128 * LDA];
  __shared__ alignas(16) u16 sAlo0[128 * LDA];
  __shared__ alignas(16) u16 sBhi0[128 * LDA];
  __shared__ alignas(16) u16 sBlo0[128 * LDA];
  __shared__ alignas(16) u16 sAhi1[128 * LDA];
  __shared__ alignas(16) u16 sAlo1[128 * LDA];
  __shared__ alignas(16) u16 sBhi1[128 * LDA];
  __shared__ alignas(16) u16 sBlo1[128 * LDA];

  const int tid   = threadIdx.x;
  const int lane  = tid & 31;
  const int wave  = tid >> 5;      // 0..7
  const int wm    = wave >> 2;     // 0..1 -> 64-row M strip
  const int wn    = wave & 3;      // 0..3 -> 32-col N strip
  const int m0    = blockIdx.y * 128;
  const int n0    = blockIdx.x * 128;
  const int row16 = lane & 15;

  v8f acc[4][2];
#pragma unroll
  for (int mt = 0; mt < 4; ++mt)
#pragma unroll
    for (int nt = 0; nt < 2; ++nt)
      acc[mt][nt] = (v8f){0.f, 0.f, 0.f, 0.f, 0.f, 0.f, 0.f, 0.f};

  // ---- A staging role: async b128 copies of prepacked bf16 planes.
  const int arow0 = tid >> 2;       // 0..63 (+64 on second copy)
  const int aj8   = (tid & 3) * 8;  // 0,8,16,24

  // ---- B staging role: thread owns tap k = tid>>3 and n = (tid&7)+8u.
  const int kl = tid >> 3;          // 0..31 (k within chunk)
  const int ng = tid & 7;           // n offset group
  int soff[16], sohw[16];           // sohw = (oh<<16)|ow, packed
#pragma unroll
  for (int u = 0; u < 16; ++u) {
    int n   = n0 + ng + 8 * u;
    int nb  = n / HW;
    int rem = n - nb * HW;
    int oh  = rem / 56;
    int ow  = rem - oh * 56;
    sohw[u] = (oh << 16) | ow;
    soff[u] = nb * CHW + oh * 56 + ow;   // packed-input offset, c=0 center tap
  }

  // Staged B registers (next chunk in flight)
  uint32_t bV[16];

  // ---- issue async A copies for chunk kkS straight into an LDS buffer
  auto stage_A_async = [&](int kkS, u16* Ahi, u16* Alo) {
    size_t g0 = (size_t)(m0 + arow0) * KTOT + kkS + aj8;
    size_t g1 = g0 + (size_t)64 * KTOT;
    async_copy_b128(lds_off32(&Ahi[arow0 * LDA + aj8]),        whi + g0);
    async_copy_b128(lds_off32(&Alo[arow0 * LDA + aj8]),        wlo + g0);
    async_copy_b128(lds_off32(&Ahi[(arow0 + 64) * LDA + aj8]), whi + g1);
    async_copy_b128(lds_off32(&Alo[(arow0 + 64) * LDA + aj8]), wlo + g1);
  };

  // ---- issue B gathers for chunk kkS into registers
  auto stage_B_regs = [&](int kkS) {
    int kg = kkS + kl;
    int c  = kg / 9;
    int rs = kg - c * 9;
    int r  = rs / 3;
    int s  = rs - r * 3;
    int dr = r - 1, dsx = s - 1;
    int cbase = c * HW + dr * 56 + dsx;
#pragma unroll
    for (int u = 0; u < 16; ++u) {
      int ih = (sohw[u] >> 16) + dr;
      int iw = (sohw[u] & 0xFFFF) + dsx;
      uint32_t v = 0u;
      if ((unsigned)ih < 56u && (unsigned)iw < 56u)
        v = xpk[(unsigned)(soff[u] + cbase)];
      bV[u] = v;
    }
  };

  // ---- write staged B registers into an LDS buffer (hi/lo peel, no VALU)
  auto stage_B_store = [&](u16* Bhi, u16* Blo) {
#pragma unroll
    for (int u = 0; u < 16; ++u) {
      int nl = ng + 8 * u;
      Bhi[nl * LDA + kl] = (u16)(bV[u] >> 16);
      Blo[nl * LDA + kl] = (u16)bV[u];
    }
  };

  // ---- 24 WMMAs from one LDS buffer
  auto compute = [&](const u16* Ahi, const u16* Alo,
                     const u16* Bhi, const u16* Blo) {
    v16bf bh[2], bl[2];
#pragma unroll
    for (int nt = 0; nt < 2; ++nt) {
      const int nrow = (wn * 32 + nt * 16 + row16) * LDA;
      bh[nt] = load_fragB(&Bhi[nrow], lane);
      bl[nt] = load_fragB(&Blo[nrow], lane);
    }
#pragma unroll
    for (int mt = 0; mt < 4; ++mt) {
      const int arow = (wm * 64 + mt * 16 + row16) * LDA;
      v16bf ah = load_fragA(&Ahi[arow], lane);
      v16bf al = load_fragA(&Alo[arow], lane);
#pragma unroll
      for (int nt = 0; nt < 2; ++nt) {
        acc[mt][nt] = __builtin_amdgcn_wmma_f32_16x16x32_bf16(
            false, ah, false, bh[nt], (short)0, acc[mt][nt], false, false);
        acc[mt][nt] = __builtin_amdgcn_wmma_f32_16x16x32_bf16(
            false, al, false, bh[nt], (short)0, acc[mt][nt], false, false);
        acc[mt][nt] = __builtin_amdgcn_wmma_f32_16x16x32_bf16(
            false, ah, false, bl[nt], (short)0, acc[mt][nt], false, false);
      }
    }
  };

  // ---- one pipeline step: async A + B loads for kk+32 -> NXT, compute kk
  //      from CUR, store B regs to NXT, wait async, single barrier.
  auto do_step = [&](int kk,
                     u16* cAhi, u16* cAlo, u16* cBhi, u16* cBlo,
                     u16* nAhi, u16* nAlo, u16* nBhi, u16* nBlo) {
    const int kn = kk + 32;
    const bool hasNext = (kn < KTOT);
    if (hasNext) {
      stage_A_async(kn, nAhi, nAlo);
      stage_B_regs(kn);
    }
    compute(cAhi, cAlo, cBhi, cBlo);
    if (hasNext) stage_B_store(nBhi, nBlo);
    wait_async0();
    __syncthreads();
  };

  // ---- prologue: stage chunk 0 into buffer 0
  stage_A_async(0, sAhi0, sAlo0);
  stage_B_regs(0);
  stage_B_store(sBhi0, sBlo0);
  wait_async0();
  __syncthreads();

  // ---- pipelined main loop; 36 chunks (even), ping-pong statically
  for (int kk = 0; kk < KTOT; kk += 64) {
    do_step(kk,      sAhi0, sAlo0, sBhi0, sBlo0, sAhi1, sAlo1, sBhi1, sBlo1);
    do_step(kk + 32, sAhi1, sAlo1, sBhi1, sBlo1, sAhi0, sAlo0, sBhi0, sBlo0);
  }

  // -------- epilogue: D layout — VGPR i: lanes0-15 (M=i,N=lane),
  // lanes16-31 (M=i+8,N=lane-16). Scatter to NCHW + bias.
  const int hilane = lane >> 4;
#pragma unroll
  for (int mt = 0; mt < 4; ++mt) {
#pragma unroll
    for (int nt = 0; nt < 2; ++nt) {
      int n   = n0 + wn * 32 + nt * 16 + row16;
      int nb  = n / HW;
      int ohw = n - nb * HW;
      int mbase = m0 + wm * 64 + mt * 16 + hilane * 8;
      float* op = out + (size_t)nb * KOHW + ohw;
#pragma unroll
      for (int i = 0; i < 8; ++i) {
        int oc = mbase + i;
        op[(size_t)oc * HW] = acc[mt][nt][i] + bias[oc];
      }
    }
  }
}

// ---------------------------------------------------------------------------
// Fallback: fused kernel (converts in-loop), used if workspace is too small.
// ---------------------------------------------------------------------------
__global__ __launch_bounds__(256)
void conv3x3_igemm_fused(const float* __restrict__ in,
                         const float* __restrict__ wgt,
                         const float* __restrict__ bias,
                         float* __restrict__ out) {
  __shared__ alignas(16) u16 sAhi[128 * LDA];
  __shared__ alignas(16) u16 sAlo[128 * LDA];
  __shared__ alignas(16) u16 sBhi[128 * LDA];
  __shared__ alignas(16) u16 sBlo[128 * LDA];

  const int tid   = threadIdx.x;
  const int lane  = tid & 31;
  const int wave  = tid >> 5;
  const int wm    = wave >> 2;
  const int wn    = wave & 3;
  const int m0    = blockIdx.y * 128;
  const int n0    = blockIdx.x * 128;
  const int row16 = lane & 15;

  v8f acc[4][2];
#pragma unroll
  for (int mt = 0; mt < 4; ++mt)
#pragma unroll
    for (int nt = 0; nt < 2; ++nt)
      acc[mt][nt] = (v8f){0.f, 0.f, 0.f, 0.f, 0.f, 0.f, 0.f, 0.f};

  const int ai = tid >> 3;
  const int aj = (tid & 7) * 4;
  const int kl = tid >> 3;
  const int ng = tid & 7;
  int soff[16], soh[16], sow[16];
#pragma unroll
  for (int u = 0; u < 16; ++u) {
    int n   = n0 + ng + 8 * u;
    int nb  = n / HW;
    int rem = n - nb * HW;
    int oh  = rem / 56;
    int ow  = rem - oh * 56;
    soh[u]  = oh;
    sow[u]  = ow;
    soff[u] = nb * CHW + oh * 56 + ow;
  }

  for (int kk = 0; kk < KTOT; kk += 32) {
#pragma unroll
    for (int rep = 0; rep < 4; ++rep) {
      int i = ai + rep * 32;
      const float4 w4 = *(const float4*)(wgt + (size_t)(m0 + i) * KTOT + kk + aj);
      float vals[4] = {w4.x, w4.y, w4.z, w4.w};
#pragma unroll
      for (int e = 0; e < 4; ++e) {
        u16 h, l;
        split_bf16(vals[e], h, l);
        sAhi[i * LDA + aj + e] = h;
        sAlo[i * LDA + aj + e] = l;
      }
    }
    {
      int kg = kk + kl;
      int c  = kg / 9;
      int rs = kg - c * 9;
      int r  = rs / 3;
      int s  = rs - r * 3;
      int dr = r - 1, dsx = s - 1;
      int cbase = c * HW + dr * 56 + dsx;
#pragma unroll
      for (int u = 0; u < 16; ++u) {
        int ih = soh[u] + dr;
        int iw = sow[u] + dsx;
        float v = 0.f;
        if ((unsigned)ih < 56u && (unsigned)iw < 56u)
          v = in[(unsigned)(soff[u] + cbase)];
        u16 h, l;
        split_bf16(v, h, l);
        int nl = ng + 8 * u;
        sBhi[nl * LDA + kl] = h;
        sBlo[nl * LDA + kl] = l;
      }
    }
    __syncthreads();

    v16bf bh[2], bl[2];
#pragma unroll
    for (int nt = 0; nt < 2; ++nt) {
      const int nrow = (wn * 32 + nt * 16 + row16) * LDA;
      bh[nt] = load_fragB(&sBhi[nrow], lane);
      bl[nt] = load_fragB(&sBlo[nrow], lane);
    }
#pragma unroll
    for (int mt = 0; mt < 4; ++mt) {
      const int arow = (wm * 64 + mt * 16 + row16) * LDA;
      v16bf ah = load_fragA(&sAhi[arow], lane);
      v16bf al = load_fragA(&sAlo[arow], lane);
#pragma unroll
      for (int nt = 0; nt < 2; ++nt) {
        acc[mt][nt] = __builtin_amdgcn_wmma_f32_16x16x32_bf16(
            false, ah, false, bh[nt], (short)0, acc[mt][nt], false, false);
        acc[mt][nt] = __builtin_amdgcn_wmma_f32_16x16x32_bf16(
            false, al, false, bh[nt], (short)0, acc[mt][nt], false, false);
        acc[mt][nt] = __builtin_amdgcn_wmma_f32_16x16x32_bf16(
            false, ah, false, bl[nt], (short)0, acc[mt][nt], false, false);
      }
    }
    __syncthreads();
  }

  const int hilane = lane >> 4;
#pragma unroll
  for (int mt = 0; mt < 4; ++mt) {
#pragma unroll
    for (int nt = 0; nt < 2; ++nt) {
      int n   = n0 + wn * 32 + nt * 16 + row16;
      int nb  = n / HW;
      int ohw = n - nb * HW;
      int mbase = m0 + wm * 64 + mt * 16 + hilane * 8;
      float* op = out + (size_t)nb * KOHW + ohw;
#pragma unroll
      for (int i = 0; i < 8; ++i) {
        int oc = mbase + i;
        op[(size_t)oc * HW] = acc[mt][nt][i] + bias[oc];
      }
    }
  }
}

extern "C" void kernel_launch(void* const* d_in, const int* in_sizes, int n_in,
                              void* d_out, int out_size, void* d_ws, size_t ws_size,
                              hipStream_t stream) {
  (void)in_sizes; (void)n_in; (void)out_size;
  const float* in   = (const float*)d_in[0];   // 32*128*56*56
  const float* wgt  = (const float*)d_in[1];   // 256*128*3*3 == [256][1152]
  const float* bias = (const float*)d_in[2];   // 256
  float* out        = (float*)d_out;           // 32*256*56*56

  // Workspace layout: whi[294912] u16 | wlo[294912] u16 | xpk[12845056] u32
  const size_t whi_off = 0;
  const size_t wlo_off = (size_t)W_ELEMS * 2;            // 589824
  const size_t xpk_off = (size_t)W_ELEMS * 4;            // 1179648
  const size_t need    = xpk_off + (size_t)IN_ELEMS * 4; // 52,559,872 B

  dim3 grid(784, 2, 1);   // N = 784*128 = 100352 ; M = 2*128 = 256
  dim3 block(256, 1, 1);

  if (ws_size >= need) {
    u16*      whi = (u16*)((char*)d_ws + whi_off);
    u16*      wlo = (u16*)((char*)d_ws + wlo_off);
    uint32_t* xpk = (uint32_t*)((char*)d_ws + xpk_off);
    prepack_weight<<<dim3(W_ELEMS / 256), block, 0, stream>>>(wgt, whi, wlo);
    prepack_input<<<dim3(IN_ELEMS / 1024), block, 0, stream>>>(in, xpk);
    conv3x3_igemm_pk<<<grid, block, 0, stream>>>(xpk, whi, wlo, bias, out);
  } else {
    conv3x3_igemm_fused<<<grid, block, 0, stream>>>(in, wgt, bias, out);
  }
}
